// LucidLinearAttention_51634096832892
// MI455X (gfx1250) — matrix operand order repaired and checked by reading, not compiled, not run on hardware
//
#include <hip/hip_runtime.h>
#include <hip/hip_bf16.h>

// ---------------------------------------------------------------------------
// LucidLinearAttention for MI455X (gfx1250, wave32, WMMA bf16 16x16x32)
//
//  1) convert x, W_qkv, W_out from f32 -> bf16 (ws)
//  2) qkv = x @ W_qkv^T  (bf16 WMMA GEMM, 128x128 macro-tile, double-buffered,
//     async global->LDS tile fill when available)
//  3) activation: q = softmax(q)/8, k = exp(k), v = v  (wave-per-group, shfl)
//  4) per-(b,h) bucketed scan: running C[64][64] (f32, LDS) and S[64];
//     per bucket: den = Q.S_prev ; attn = (Q @ C_prev) * Dinv  (WMMA)
//                 C  += K^T @ V (WMMA) ; S += colsum(K)
//  5) out = xo @ W_out^T (same GEMM, f32 output)
// ---------------------------------------------------------------------------

#define BB      4
#define TT      4096
#define DM      1024
#define NH      16
#define HD      64
#define NBUCKET 64
#define BK      64
#define MTOT    (BB * TT)          // 16384 rows

typedef __attribute__((ext_vector_type(16))) __bf16 v16bf;
typedef __attribute__((ext_vector_type(8)))  float  v8f;

static __device__ __forceinline__ float bf2f(__bf16 h) {
  unsigned short u = __builtin_bit_cast(unsigned short, h);
  unsigned v = ((unsigned)u) << 16;
  return __builtin_bit_cast(float, v);
}
static __device__ __forceinline__ __bf16 f2bf(float f) {
  unsigned u = __builtin_bit_cast(unsigned, f);
  u = u + 0x7FFFu + ((u >> 16) & 1u);               // round-to-nearest-even
  unsigned short s = (unsigned short)(u >> 16);
  return __builtin_bit_cast(__bf16, s);
}

// ---- async global -> LDS (GLOBAL_LOAD_ASYNC_TO_LDS_B128, ASYNCcnt) ---------
#if __has_builtin(__builtin_amdgcn_global_load_async_to_lds_b128)
#define ASYNC_G2L 1
typedef __attribute__((vector_size(16))) int vi4;
typedef __attribute__((address_space(1))) vi4 as1_vi4;   // global (__device__)
typedef __attribute__((address_space(3))) vi4 as3_vi4;   // LDS (__shared__)
static __device__ __forceinline__ void g2l_b128(void* lds, const void* g) {
  // generic LDS pointer keeps the LDS byte offset in addr[31:0] (ISA 10.2)
  __builtin_amdgcn_global_load_async_to_lds_b128(
      (as1_vi4*)(unsigned long long)(size_t)g,
      (as3_vi4*)(unsigned)(size_t)lds, 0, 0);
}
static __device__ __forceinline__ void async_wait0() {
#if __has_builtin(__builtin_amdgcn_s_wait_asynccnt)
  __builtin_amdgcn_s_wait_asynccnt(0);
#else
  asm volatile("s_wait_asynccnt 0" ::: "memory");
#endif
}
#else
static __device__ __forceinline__ void g2l_b128(void* lds, const void* g) {
  *(int4*)lds = *(const int4*)g;
}
static __device__ __forceinline__ void async_wait0() {}
#endif

// ---- WMMA fragment builders (contiguous k = 16*(lane/16) + s for A and B, wave32) -------------------
// A 16x32 bf16: lane<16 -> row M=lane, K {0..15}; lane>=16 -> row M=lane-16, K {16..31}
static __device__ __forceinline__ v16bf frag_A_rowmajor(const __bf16* S, int ld,
                                                        int m0, int k0, int lane) {
  const __bf16* p = S + (m0 + (lane & 15)) * ld + k0 + ((lane >> 4) ? 16 : 0);
  v16bf a;
#pragma unroll
  for (int j = 0; j < 8; ++j) a[j] = p[j];
#pragma unroll
  for (int j = 0; j < 8; ++j) a[8 + j] = p[8 + j];
  return a;
}
// A 16x32 from transposed storage: want A[m=d][k=s], storage S[s][d]
static __device__ __forceinline__ v16bf frag_A_transposed(const __bf16* S, int ld,
                                                          int d0, int s0, int lane) {
  int d = d0 + (lane & 15);
  int sb = s0 + ((lane >> 4) ? 16 : 0);
  v16bf a;
#pragma unroll
  for (int j = 0; j < 8; ++j) a[j] = S[(sb + j) * ld + d];
#pragma unroll
  for (int j = 0; j < 8; ++j) a[8 + j] = S[(sb + 8 + j) * ld + d];
  return a;
}
// B 32x16 where source is row-major [N][K] (weight rows): lane = col N, halves split K
static __device__ __forceinline__ v16bf frag_B_NK(const __bf16* S, int ld,
                                                  int n0, int k0, int lane) {
  const __bf16* p = S + (n0 + (lane & 15)) * ld + k0 + ((lane >> 4) ? 16 : 0);
  v16bf b;
#pragma unroll
  for (int j = 0; j < 16; ++j) b[j] = p[j];
  return b;
}
// B 32x16 where source is [K][N] bf16
static __device__ __forceinline__ v16bf frag_B_KN(const __bf16* S, int ld,
                                                  int k0, int n0, int lane) {
  int n = n0 + (lane & 15);
  int kb = k0 + ((lane >> 4) ? 16 : 0);
  v16bf b;
#pragma unroll
  for (int j = 0; j < 16; ++j) b[j] = S[(kb + j) * ld + n];
  return b;
}
// B 32x16 where source is [K][N] f32 (running context C), converted on read
static __device__ __forceinline__ v16bf frag_B_KN_f32(const float* S, int ld,
                                                      int k0, int n0, int lane) {
  int n = n0 + (lane & 15);
  int kb = k0 + ((lane >> 4) ? 16 : 0);
  v16bf b;
#pragma unroll
  for (int j = 0; j < 16; ++j) b[j] = f2bf(S[(kb + j) * ld + n]);
  return b;
}

static __device__ __forceinline__ v8f wmma_bf16(v16bf a, v16bf b, v8f c) {
  return __builtin_amdgcn_wmma_f32_16x16x32_bf16(false, a, false, b,
                                                 (short)0, c, false, false);
}

// ---------------------------------------------------------------------------
// 1) f32 -> bf16 conversion
__global__ void cvt_f32_bf16(const float* __restrict__ src,
                             __bf16* __restrict__ dst, long n) {
  long i = (long)blockIdx.x * blockDim.x + threadIdx.x;
  long stride = (long)gridDim.x * blockDim.x;
  for (; i < n; i += stride) dst[i] = f2bf(src[i]);
}

// ---------------------------------------------------------------------------
// 2)/5) Tiled bf16 GEMM: C[M,N] = A[M,K] @ B[N,K]^T
//   128x128 macro-tile, 256 threads (8 waves), wave = 64x32 sub-block
//   => 8 WMMAs per wave per 32-deep K-step; double-buffered LDS (one barrier/step)
__global__ __launch_bounds__(256) void gemm_bf16(const __bf16* __restrict__ A,
                                                 const __bf16* __restrict__ B,
                                                 float* __restrict__ Cf,
                                                 __bf16* __restrict__ Cb,
                                                 int M, int N, int K, int out_bf) {
  __shared__ __align__(16) __bf16 As[2][128][32];
  __shared__ __align__(16) __bf16 Bs[2][128][32];

  const int m0 = blockIdx.y * 128;
  const int n0 = blockIdx.x * 128;
  const int tid = threadIdx.x;
  const int lane = tid & 31;
  const int wave = tid >> 5;
  const int wm = (wave >> 2) * 64;   // wave M offset within macro-tile: 0 / 64
  const int wn = (wave & 3) * 32;    // wave N offset within macro-tile: 0/32/64/96

  // loader: each thread brings 16 bf16 (32B) per tile per K-step
  const int lrow = tid >> 1;         // 0..127
  const int lcol = (tid & 1) * 16;   // 0 / 16

  v8f acc[4][2] = {};

  const __bf16* pa = &A[(size_t)(m0 + lrow) * K + lcol];
  const __bf16* pb = &B[(size_t)(n0 + lrow) * K + lcol];

#define LOAD_TILES(k0, buf)                                                   \
  do {                                                                        \
    g2l_b128(&As[buf][lrow][lcol],     pa + (k0));                            \
    g2l_b128(&As[buf][lrow][lcol + 8], pa + (k0) + 8);                        \
    g2l_b128(&Bs[buf][lrow][lcol],     pb + (k0));                            \
    g2l_b128(&Bs[buf][lrow][lcol + 8], pb + (k0) + 8);                        \
  } while (0)

  LOAD_TILES(0, 0);
  async_wait0();
  __syncthreads();

  int buf = 0;
  for (int k0 = 0; k0 < K; k0 += 32) {
    if (k0 + 32 < K) {
      LOAD_TILES(k0 + 32, buf ^ 1);            // async fill overlaps the WMMAs
      __builtin_prefetch(pa + k0 + 64, 0, 1);  // global_prefetch_b8 for K-step+2
      __builtin_prefetch(pb + k0 + 64, 0, 1);
    }
    v16bf af[4], bfv[2];
#pragma unroll
    for (int mi = 0; mi < 4; ++mi)
      af[mi] = frag_A_rowmajor(&As[buf][0][0], 32, wm + mi * 16, 0, lane);
#pragma unroll
    for (int ni = 0; ni < 2; ++ni)
      bfv[ni] = frag_B_NK(&Bs[buf][0][0], 32, wn + ni * 16, 0, lane);
#pragma unroll
    for (int mi = 0; mi < 4; ++mi)
#pragma unroll
      for (int ni = 0; ni < 2; ++ni)
        acc[mi][ni] = wmma_bf16(af[mi], bfv[ni], acc[mi][ni]);
    async_wait0();                             // own async fills landed
    __syncthreads();                           // everyone's fills visible
    buf ^= 1;
  }
#undef LOAD_TILES

  // D layout: lane<16 -> N=lane, M rows +r ; lane>=16 -> N=lane-16, M rows +8+r
  const int mhalf = ((lane >> 4) ? 8 : 0);
  const int ncl = (lane & 15);
#pragma unroll
  for (int mi = 0; mi < 4; ++mi) {
#pragma unroll
    for (int ni = 0; ni < 2; ++ni) {
      const int mb = m0 + wm + mi * 16 + mhalf;
      const int nc = n0 + wn + ni * 16 + ncl;
#pragma unroll
      for (int r = 0; r < 8; ++r) {
        size_t o = (size_t)(mb + r) * N + nc;
        if (out_bf) Cb[o] = f2bf(acc[mi][ni][r]);
        else        Cf[o] = acc[mi][ni][r];
      }
    }
  }
}

// ---------------------------------------------------------------------------
// 3) activation: one wave32 per (t,h) group of 64 channels; shfl_xor reductions
__global__ __launch_bounds__(256) void qkv_act(const __bf16* __restrict__ qkv,
                                               __bf16* __restrict__ q,
                                               __bf16* __restrict__ k,
                                               __bf16* __restrict__ v) {
  const int wid = blockIdx.x * 8 + (threadIdx.x >> 5);  // group = t*NH + h
  if (wid >= MTOT * NH) return;
  const int lane = threadIdx.x & 31;
  const int t = wid >> 4, h = wid & 15;
  const __bf16* base = qkv + (size_t)t * (3 * DM) + h * HD;
  const size_t dbase = (size_t)t * DM + h * HD;

  // q: softmax over 64 channels (2 per lane), * HD^-0.5 (=1/8)
  float x0 = bf2f(base[lane]);
  float x1 = bf2f(base[lane + 32]);
  float mx = fmaxf(x0, x1);
#pragma unroll
  for (int off = 16; off > 0; off >>= 1) mx = fmaxf(mx, __shfl_xor(mx, off, 32));
  float e0 = __expf(x0 - mx), e1 = __expf(x1 - mx);
  float s = e0 + e1;
#pragma unroll
  for (int off = 16; off > 0; off >>= 1) s += __shfl_xor(s, off, 32);
  const float inv = 0.125f / s;
  q[dbase + lane]      = f2bf(e0 * inv);
  q[dbase + lane + 32] = f2bf(e1 * inv);
  // k: exp
  k[dbase + lane]      = f2bf(__expf(bf2f(base[DM + lane])));
  k[dbase + lane + 32] = f2bf(__expf(bf2f(base[DM + lane + 32])));
  // v: copy
  v[dbase + lane]      = base[2 * DM + lane];
  v[dbase + lane + 32] = base[2 * DM + lane + 32];
}

// ---------------------------------------------------------------------------
// 4) bucketed linear-attention scan, one block per (b,h)
__global__ __launch_bounds__(256) void lucid_scan(const __bf16* __restrict__ qb,
                                                  const __bf16* __restrict__ kb,
                                                  const __bf16* __restrict__ vb,
                                                  __bf16* __restrict__ xo) {
  __shared__ __align__(16) __bf16 Qs[BK][HD];
  __shared__ __align__(16) __bf16 Ks[BK][HD];
  __shared__ __align__(16) __bf16 Vs[BK][HD];
  __shared__ float Cs[HD][HD];      // running context (exclusive)
  __shared__ float Ssum[HD];        // running key-sum (exclusive)
  __shared__ float Dinv[BK];

  const int bh = blockIdx.x;        // 0..63
  const int b = bh >> 4, h = bh & 15;
  const int tid = threadIdx.x;
  const int lane = tid & 31;
  const int wave = tid >> 5;
  const int mi = wave & 3;
  const int nj = (wave >> 2) * 2;

  for (int i = tid; i < HD * HD; i += 256) (&Cs[0][0])[i] = 0.f;
  if (tid < HD) Ssum[tid] = 0.f;
  __syncthreads();

  const int lrow = tid >> 2;        // 0..63 (token s in bucket)
  const int lcol = (tid & 3) * 16;  // 0,16,32,48

  for (int n = 0; n < NBUCKET; ++n) {
    const size_t t0 = (size_t)b * TT + (size_t)n * BK;
    const size_t src = (t0 + lrow) * DM + (size_t)h * HD + lcol;
    g2l_b128(&Qs[lrow][lcol],     &qb[src]);
    g2l_b128(&Qs[lrow][lcol + 8], &qb[src + 8]);
    g2l_b128(&Ks[lrow][lcol],     &kb[src]);
    g2l_b128(&Ks[lrow][lcol + 8], &kb[src + 8]);
    g2l_b128(&Vs[lrow][lcol],     &vb[src]);
    g2l_b128(&Vs[lrow][lcol + 8], &vb[src + 8]);
    async_wait0();
    __syncthreads();

    // den[s] = q[s,:] . S_prev ; Dinv = 1/max(den,1e-6)
    if (tid < BK) {
      float acc = 0.f;
#pragma unroll 4
      for (int d = 0; d < HD; ++d) acc += bf2f(Qs[tid][d]) * Ssum[d];
      Dinv[tid] = 1.0f / fmaxf(acc, 1e-6f);
    }
    __syncthreads();

    // attn = (Q @ C_prev) * Dinv[row] -> xo (bf16, cat-heads layout)
#pragma unroll
    for (int jj = 0; jj < 2; ++jj) {
      const int ni = nj + jj;
      v8f acc = {};
#pragma unroll
      for (int kk = 0; kk < HD; kk += 32) {
        v16bf a = frag_A_rowmajor(&Qs[0][0], HD, mi * 16, kk, lane);
        v16bf bfm = frag_B_KN_f32(&Cs[0][0], HD, kk, ni * 16, lane);
        acc = wmma_bf16(a, bfm, acc);
      }
      const int e = ni * 16 + (lane & 15);
      const int mb = mi * 16 + ((lane >> 4) ? 8 : 0);
#pragma unroll
      for (int r = 0; r < 8; ++r) {
        const int m = mb + r;
        xo[(t0 + m) * DM + (size_t)h * HD + e] = f2bf(acc[r] * Dinv[m]);
      }
    }
    __syncthreads();   // all reads of Cs / Ssum done before updating them

    // S += colsum(K)
    if (tid < HD) {
      float acc = 0.f;
#pragma unroll 4
      for (int s = 0; s < BK; ++s) acc += bf2f(Ks[s][tid]);
      Ssum[tid] += acc;
    }
    // C += K^T @ V
#pragma unroll
    for (int jj = 0; jj < 2; ++jj) {
      const int ei = nj + jj;
      v8f acc = {};
#pragma unroll
      for (int kk = 0; kk < BK; kk += 32) {
        v16bf a = frag_A_transposed(&Ks[0][0], HD, mi * 16, kk, lane);
        v16bf bfm = frag_B_KN(&Vs[0][0], HD, kk, ei * 16, lane);
        acc = wmma_bf16(a, bfm, acc);
      }
      const int e = ei * 16 + (lane & 15);
      const int mb = mi * 16 + ((lane >> 4) ? 8 : 0);
#pragma unroll
      for (int r = 0; r < 8; ++r) Cs[mb + r][e] += acc[r];
    }
    __syncthreads();   // updates visible; safe to overwrite Q/K/V tiles
  }
}

// ---------------------------------------------------------------------------
extern "C" void kernel_launch(void* const* d_in, const int* in_sizes, int n_in,
                              void* d_out, int out_size, void* d_ws, size_t ws_size,
                              hipStream_t stream) {
  (void)in_sizes; (void)n_in; (void)out_size; (void)ws_size;
  const float* x     = (const float*)d_in[0];   // [4,4096,1024]
  const float* Wqkv  = (const float*)d_in[1];   // [3072,1024]
  const float* Wout  = (const float*)d_in[2];   // [1024,1024]
  float* out = (float*)d_out;                   // [4,4096,1024]

  char* ws = (char*)d_ws;
  const size_t n_x    = (size_t)MTOT * DM;          // 16.7M
  const size_t n_wqkv = (size_t)3 * DM * DM;        // 3.1M
  const size_t n_wout = (size_t)DM * DM;            // 1.0M
  const size_t n_qkv  = (size_t)MTOT * 3 * DM;      // 50.3M

  __bf16* xbf    = (__bf16*)ws;                 ws += n_x    * 2;
  __bf16* wqkvbf = (__bf16*)ws;                 ws += n_wqkv * 2;
  __bf16* woutbf = (__bf16*)ws;                 ws += n_wout * 2;
  __bf16* qkvbf  = (__bf16*)ws;                 ws += n_qkv  * 2;
  __bf16* q      = (__bf16*)ws;                 ws += n_x * 2;
  __bf16* k      = (__bf16*)ws;                 ws += n_x * 2;
  __bf16* v      = (__bf16*)ws;                 ws += n_x * 2;
  __bf16* xobf   = (__bf16*)ws;                 ws += n_x * 2;

  // 1) conversions
  cvt_f32_bf16<<<4096, 256, 0, stream>>>(x, xbf, (long)n_x);
  cvt_f32_bf16<<<2048, 256, 0, stream>>>(Wqkv, wqkvbf, (long)n_wqkv);
  cvt_f32_bf16<<<1024, 256, 0, stream>>>(Wout, woutbf, (long)n_wout);

  // 2) qkv = x @ Wqkv^T   (M=16384, N=3072, K=1024) -> bf16
  gemm_bf16<<<dim3(3 * DM / 128, MTOT / 128), 256, 0, stream>>>(
      xbf, wqkvbf, nullptr, qkvbf, MTOT, 3 * DM, DM, 1);

  // 3) activations (one wave per (t,h) group; 8 waves/block)
  qkv_act<<<(MTOT * NH) / 8, 256, 0, stream>>>(qkvbf, q, k, v);

  // 4) bucketed scan, one block per (b,h)
  lucid_scan<<<BB * NH, 256, 0, stream>>>(q, k, v, xobf);

  // 5) out = xo @ Wout^T  (M=16384, N=1024, K=1024) -> f32
  gemm_bf16<<<dim3(DM / 128, MTOT / 128), 256, 0, stream>>>(
      xobf, woutbf, out, nullptr, MTOT, DM, DM, 0);
}
